// GCNConvEmbedder_11467562680659
// MI455X (gfx1250) — matrix operand order repaired and checked by reading, not compiled, not run on hardware
//
#include <hip/hip_runtime.h>

// ---------------------------------------------------------------------------
// GraphSAGE (3 layers, D=64) + mean pooling for MI455X (gfx1250, wave32).
//   layer: agg = segment_mean(x[src], dst);  x' = agg@Wl + bl + x@Wr
// Fused GEMM: [agg/deg | x] (N x 128) @ [Wl ; Wr] (128 x 64) via
// v_wmma_f32_16x16x32_f16, one 16x16 tile per wave, K=128 = 4 WMMAs.
// Weights pre-swizzled to fragment layout once per layer, then copied to LDS
// per block with global_load_async_to_lds_b128 (+ s_wait_asynccnt).
// Edge aggregation via global_atomic_add_f32 (node table is L2-resident).
// ---------------------------------------------------------------------------

typedef __attribute__((ext_vector_type(16))) _Float16 v16h;
typedef __attribute__((ext_vector_type(8)))  _Float16 v8h;
typedef __attribute__((ext_vector_type(8)))  float    v8f;

#define D_FEAT 64

// ---------------------------- zero scratch ---------------------------------
__global__ void zero_f32(float* __restrict__ p, int n) {
    int i = blockIdx.x * blockDim.x + threadIdx.x;
    int stride = gridDim.x * blockDim.x;
    for (; i < n; i += stride) p[i] = 0.0f;
}

// ------------------------- edge scatter (mean agg) -------------------------
// One thread per (edge, 4-float chunk): 16 threads cover a 64-float row.
__global__ void edge_agg(const float* __restrict__ x,
                         const int* __restrict__ src,
                         const int* __restrict__ dst,
                         float* __restrict__ agg,
                         float* __restrict__ deg,
                         int n_edges) {
    int idx = blockIdx.x * blockDim.x + threadIdx.x;
    int e = idx >> 4;
    int q = idx & 15;
    if (e >= n_edges) return;
    int s = src[e];
    int d = dst[e];
    // prefetch a future edge's source row (lowers to global_prefetch_b8)
    if (e + 8 < n_edges) {
        int sf = src[e + 8];
        __builtin_prefetch(x + (size_t)sf * D_FEAT + q * 4, 0, 1);
    }
    const float4 v = ((const float4*)(x + (size_t)s * D_FEAT))[q];
    float* dp = agg + (size_t)d * D_FEAT + q * 4;
    atomicAdd(dp + 0, v.x);
    atomicAdd(dp + 1, v.y);
    atomicAdd(dp + 2, v.z);
    atomicAdd(dp + 3, v.w);
    if (q == 0) atomicAdd(deg + d, 1.0f);
}

// --------------- weight pre-swizzle: [Wl;Wr] -> B-fragment f16 -------------
// wfrag index = ((colT*4 + kb)*32 + lane)*16 + i, so each lane's 16-element
// B fragment is 32 contiguous bytes. B layout per ISA: lanes 0-15 hold
// K = kb*32 + 0..7,16..23 (2 halves/VGPR), lanes 16-31 hold K = kb*32 + 8..15,24..31,
// column N = colT*16 + (lane&15).
__global__ void prep_wfrag(const float* __restrict__ Wl,
                           const float* __restrict__ Wr,
                           _Float16* __restrict__ wfrag) {
    int j = blockIdx.x * blockDim.x + threadIdx.x;   // 0 .. 8191
    if (j >= 4 * 4 * 32 * 16) return;
    int i    = j & 15;
    int lane = (j >> 4) & 31;
    int kb   = (j >> 9) & 3;
    int colT = j >> 11;
    int lrow = lane & 15;
    int lhi  = lane >> 4;
    int k    = kb * 32 + (i >> 3) * 16 + lhi * 8 + (i & 7); // 0 .. 127
    int n    = colT * 16 + lrow;                     // 0 .. 63
    float w = (k < 64) ? Wl[k * 64 + n] : Wr[(k - 64) * 64 + n];
    wfrag[j] = (_Float16)w;
}

// ---------------------- fused SAGE linear (WMMA GEMM) ----------------------
// Block = 256 threads = 8 waves; computes 32 rows x 64 cols of the output.
// A = [agg/deg | x] staged in LDS as f16 (32 x 128); B = pre-swizzled weight
// fragments (16 KB) async-copied into LDS. Wave w -> rowTile w/4, colTile w%4.
__global__ void sage_linear(const float* __restrict__ xin,
                            const float* __restrict__ agg,
                            const float* __restrict__ deg,
                            const _Float16* __restrict__ wfrag, // [8192]
                            const float* __restrict__ bl,       // [64]
                            float* __restrict__ xout,
                            int n_nodes) {
    __shared__ _Float16 s_rows[32 * 128];     //  8 KB: [row][k]
    __shared__ _Float16 s_w[4 * 4 * 32 * 16]; // 16 KB fragment-major weights

    const int tid = threadIdx.x;
    const int row0 = blockIdx.x * 32;

    // ---- async-copy weights global -> LDS (1024 x 16B, 4 per thread)
    {
        unsigned lds0 = (unsigned)(unsigned long long)(const void*)&s_w[0];
        #pragma unroll
        for (int it = 0; it < 4; ++it) {
            int j = tid + it * 256;
            unsigned ldsa = lds0 + (unsigned)(j * 16);
            unsigned long long ga =
                (unsigned long long)(const void*)(wfrag + (size_t)j * 8);
            asm volatile("global_load_async_to_lds_b128 %0, %1, off"
                         :: "v"(ldsa), "v"(ga)
                         : "memory");
        }
    }

    // ---- stage A rows (f32 -> f16, packed 128-bit LDS stores)
    {
        int r   = tid >> 3;       // 0..31
        int seg = tid & 7;        // 0..7 -> cols seg*16 .. seg*16+15
        int node = row0 + r;
        v8h* dstp = (v8h*)(s_rows + r * 128 + seg * 16);
        v8h lo = (v8h)(_Float16)0.0f, hi = (v8h)(_Float16)0.0f;
        if (node < n_nodes) {
            float scale;
            const float* sp;
            if (seg < 4) {
                scale = 1.0f / fmaxf(deg[node], 1.0f);
                sp = agg + (size_t)node * D_FEAT + seg * 16;
            } else {
                scale = 1.0f;
                sp = xin + (size_t)node * D_FEAT + (seg - 4) * 16;
            }
            const float4* sp4 = (const float4*)sp;
            float4 f0 = sp4[0], f1 = sp4[1], f2 = sp4[2], f3 = sp4[3];
            lo = (v8h){(_Float16)(f0.x * scale), (_Float16)(f0.y * scale),
                       (_Float16)(f0.z * scale), (_Float16)(f0.w * scale),
                       (_Float16)(f1.x * scale), (_Float16)(f1.y * scale),
                       (_Float16)(f1.z * scale), (_Float16)(f1.w * scale)};
            hi = (v8h){(_Float16)(f2.x * scale), (_Float16)(f2.y * scale),
                       (_Float16)(f2.z * scale), (_Float16)(f2.w * scale),
                       (_Float16)(f3.x * scale), (_Float16)(f3.y * scale),
                       (_Float16)(f3.z * scale), (_Float16)(f3.w * scale)};
        }
        dstp[0] = lo;
        dstp[1] = hi;
    }

    asm volatile("s_wait_asynccnt 0x0" ::: "memory");
    __syncthreads();

    const int lane = tid & 31;
    const int wv   = tid >> 5;
    const int rowT = wv >> 2;     // 0..1
    const int colT = wv & 3;      // 0..3
    const int lrow = lane & 15;   // row (A) / column (B,C) within tile
    const int lhi  = lane >> 4;   // half-wave selector per ISA layouts

    const _Float16* arow = s_rows + (rowT * 16 + lrow) * 128;
    const _Float16* bfr  = s_w + ((colT * 4) * 32 + lane) * 16;

    v8f acc = {};
    #pragma unroll
    for (int kb = 0; kb < 4; ++kb) {   // K = 128 in 4 chunks of 32
        // A fragment (16x32 f16): two contiguous 8-half runs per lane
        // (K = kb*32 + lhi*8 + 0..7 and +16): two ds_load_b128.
        v8h a0 = *(const v8h*)(arow + kb * 32 + lhi * 8);
        v8h a1 = *(const v8h*)(arow + kb * 32 + 16 + lhi * 8);
        v16h a = __builtin_shufflevector(a0, a1, 0, 1, 2, 3, 4, 5, 6, 7,
                                         8, 9, 10, 11, 12, 13, 14, 15);
        // B fragment: pre-swizzled, 16 contiguous halves per lane.
        const v8h* bp = (const v8h*)(bfr + kb * 32 * 16);
        v8h b0 = bp[0];
        v8h b1 = bp[1];
        v16h b = __builtin_shufflevector(b0, b1, 0, 1, 2, 3, 4, 5, 6, 7,
                                         8, 9, 10, 11, 12, 13, 14, 15);
        acc = __builtin_amdgcn_wmma_f32_16x16x32_f16(
            /*neg_a=*/false, a, /*neg_b=*/false, b,
            /*c_mod=*/(short)0, acc, /*reuse_a=*/false, /*reuse_b=*/false);
    }

    // ---- bias (per-column broadcast) + store. C/D layout: VGPR r holds
    // row M = r + 8*lhi, column N = lane&15. Fast path: one base pointer,
    // 8 immediate-offset stores, no per-row guards (wave-uniform branch).
    const int gcol = colT * 16 + lrow;
    const float bias = bl[gcol];
    const int baseRow = row0 + rowT * 16 + lhi * 8;
    float* op = xout + (size_t)baseRow * D_FEAT + gcol;
    if (row0 + 32 <= n_nodes) {
        #pragma unroll
        for (int r = 0; r < 8; ++r) op[r * D_FEAT] = acc[r] + bias;
    } else {
        #pragma unroll
        for (int r = 0; r < 8; ++r)
            if (baseRow + r < n_nodes) op[r * D_FEAT] = acc[r] + bias;
    }
}

// ----------------------------- graph pooling -------------------------------
__global__ void pool_accum(const float* __restrict__ x,
                           const int* __restrict__ batch,
                           float* __restrict__ psum,
                           float* __restrict__ pcnt,
                           int n_nodes) {
    int idx = blockIdx.x * blockDim.x + threadIdx.x;
    int node = idx >> 6;
    int f = idx & 63;
    if (node >= n_nodes) return;
    int g = batch[node];
    atomicAdd(psum + (size_t)g * D_FEAT + f, x[(size_t)node * D_FEAT + f]);
    if (f == 0) atomicAdd(pcnt + g, 1.0f);
}

__global__ void pool_final(const float* __restrict__ psum,
                           const float* __restrict__ pcnt,
                           float* __restrict__ out,
                           int total) {
    int idx = blockIdx.x * blockDim.x + threadIdx.x;
    if (idx >= total) return;
    int g = idx >> 6;
    out[idx] = psum[idx] / fmaxf(pcnt[g], 1.0f);
}

// ------------------------------- launcher ----------------------------------
extern "C" void kernel_launch(void* const* d_in, const int* in_sizes, int n_in,
                              void* d_out, int out_size, void* d_ws, size_t ws_size,
                              hipStream_t stream) {
    const float* x   = (const float*)d_in[0];   // [N, 64] f32
    const int*   ei  = (const int*)d_in[1];     // [2, E]
    const int*   bat = (const int*)d_in[2];     // [N]
    const float* Wl  = (const float*)d_in[3];   // [3, 64, 64]
    const float* bl  = (const float*)d_in[4];   // [3, 64]
    const float* Wr  = (const float*)d_in[5];   // [3, 64, 64]

    const int E = in_sizes[1] / 2;
    const int N = in_sizes[2];
    const int G = out_size / D_FEAT;            // 64 graphs

    // workspace layout
    float* xa   = (float*)d_ws;                 // [N, 64]
    float* xb   = xa + (size_t)N * D_FEAT;      // [N, 64]
    float* agg  = xb + (size_t)N * D_FEAT;      // [N, 64]
    float* deg  = agg + (size_t)N * D_FEAT;     // [N]
    float* psum = deg + N;                      // [G, 64]
    float* pcnt = psum + (size_t)G * D_FEAT;    // [G]
    _Float16* wfrag = (_Float16*)(pcnt + G);    // [8192] swizzled f16 weights

    const int BS = 256;
    const int zeroGrid = 2048;
    const int edgeGrid = (E * 16 + BS - 1) / BS;
    const int gemmGrid = (N + 31) / 32;
    const int poolGrid = (N * D_FEAT + BS - 1) / BS;

    const float* cur = x;
    float* nxt = xa;
    for (int l = 0; l < 3; ++l) {
        zero_f32<<<zeroGrid, BS, 0, stream>>>(agg, N * D_FEAT);
        zero_f32<<<64, BS, 0, stream>>>(deg, N);
        edge_agg<<<edgeGrid, BS, 0, stream>>>(cur, ei, ei + E, agg, deg, E);
        prep_wfrag<<<32, BS, 0, stream>>>(Wl + l * 64 * 64, Wr + l * 64 * 64,
                                          wfrag);
        sage_linear<<<gemmGrid, BS, 0, stream>>>(cur, agg, deg, wfrag,
                                                 bl + l * 64, nxt, N);
        cur = nxt;
        nxt = (l == 0) ? xb : xa;
    }

    zero_f32<<<16, BS, 0, stream>>>(psum, G * D_FEAT);
    zero_f32<<<1, BS, 0, stream>>>(pcnt, G);
    pool_accum<<<poolGrid, BS, 0, stream>>>(cur, bat, psum, pcnt, N);
    pool_final<<<(G * D_FEAT + BS - 1) / BS, BS, 0, stream>>>(
        psum, pcnt, (float*)d_out, G * D_FEAT);
}
